// edge_graph_pool_22737556865355
// MI455X (gfx1250) — matrix operand order repaired and checked by reading, not compiled, not run on hardware
//
#include <hip/hip_runtime.h>
#include <hip/hip_bf16.h>

// ---------------------------------------------------------------------------
// Problem constants (from the reference)
// ---------------------------------------------------------------------------
#define EDGES 500000
#define DDIM  512
#define SEGS  50000
#define LNEPS 1e-5f

typedef __attribute__((ext_vector_type(16))) __bf16 v16bf;
typedef __attribute__((ext_vector_type(8)))  float  v8f;

// ---------------------------------------------------------------------------
// bf16 helpers (round-to-nearest-even), used for the hi/lo compensated split:
//   x = hi + lo  with hi = bf16(x), lo = bf16(x - hi)
// => A*B = Ahi*Bhi + Ahi*Blo + Alo*Bhi (+ O(2^-16) dropped term)
// ---------------------------------------------------------------------------
__device__ __forceinline__ unsigned short f32_to_bf16_rne(float x) {
  unsigned int u = __float_as_uint(x);
  unsigned int r = u + 0x7FFFu + ((u >> 16) & 1u);
  return (unsigned short)(r >> 16);
}
__device__ __forceinline__ float bf16_to_f32(unsigned short h) {
  return __uint_as_float(((unsigned int)h) << 16);
}

// ---------------------------------------------------------------------------
// Kernel 1: zero the scatter accumulators (ws is poisoned, not re-zeroed)
// ---------------------------------------------------------------------------
__global__ void zero_kernel(float* __restrict__ sums, float* __restrict__ cnt) {
  long long i = (long long)blockIdx.x * blockDim.x + threadIdx.x;
  const long long SD = (long long)SEGS * DDIM;
  if (i < SD) sums[i] = 0.0f;
  else if (i < SD + SEGS) cnt[i - SD] = 0.0f;
}

// ---------------------------------------------------------------------------
// Kernel 2: scatter-add (segment sums + counts). 128 threads per edge row,
// float4 loads (coalesced), HW fp32 atomics resolve in L2 (sums fit in 192MB).
// ---------------------------------------------------------------------------
__global__ void scatter_kernel(const float* __restrict__ edge_graph,
                               const int* __restrict__ edge_index,
                               float* __restrict__ sums,
                               float* __restrict__ cnt) {
  long long tid = (long long)blockIdx.x * blockDim.x + threadIdx.x;
  long long e = tid >> 7;          // edge id
  int q = (int)(tid & 127);        // float4 chunk within the 512-wide row
  if (e >= EDGES) return;
  int seg = edge_index[EDGES + e]; // row 1 of edge_index (int32: JAX x64 off)
  float4 v = reinterpret_cast<const float4*>(edge_graph)[e * 128 + q];
  float* dst = sums + (long long)seg * DDIM + q * 4;
  unsafeAtomicAdd(dst + 0, v.x);
  unsafeAtomicAdd(dst + 1, v.y);
  unsafeAtomicAdd(dst + 2, v.z);
  unsafeAtomicAdd(dst + 3, v.w);
  if (q == 0) unsafeAtomicAdd(&cnt[seg], 1.0f);
}

// ---------------------------------------------------------------------------
// Kernel 3: pack W (row-major [j,k]) into a FRAGMENT-MAJOR transposed layout:
// for K-tile kt (32 wide), column j, the 16 packed K-pairs live contiguously:
//   Bf[(kt*512 + j)*16 + p] = pack(bf16(W[j, kt*32+2p]), bf16(W[j, kt*32+2p+1]))
// so a lane's B fragment half (8 dwords) is 32 contiguous bytes -> 2x b128.
// ---------------------------------------------------------------------------
__global__ void pack_w_kernel(const float* __restrict__ W,
                              unsigned int* __restrict__ Bfhi,
                              unsigned int* __restrict__ Bflo) {
  int i = blockIdx.x * blockDim.x + threadIdx.x; // 0 .. 256*512-1
  if (i >= (DDIM / 2) * DDIM) return;
  int kk = i >> 9;        // global k-pair index 0..255
  int j  = i & (DDIM - 1);
  int k  = kk * 2;
  float w0 = W[(long long)j * DDIM + k];
  float w1 = W[(long long)j * DDIM + k + 1];
  unsigned short h0 = f32_to_bf16_rne(w0);
  unsigned short h1 = f32_to_bf16_rne(w1);
  unsigned short l0 = f32_to_bf16_rne(w0 - bf16_to_f32(h0));
  unsigned short l1 = f32_to_bf16_rne(w1 - bf16_to_f32(h1));
  int kt = kk >> 4;       // K-tile index (32 elements per tile)
  int p  = kk & 15;       // pair-within-tile == half*8 + v
  long long dst = ((long long)kt * DDIM + j) * 16 + p;
  Bfhi[dst] = (unsigned int)h0 | ((unsigned int)h1 << 16);
  Bflo[dst] = (unsigned int)l0 | ((unsigned int)l1 << 16);
}

// ---------------------------------------------------------------------------
// Kernel 4: mean (sums/max(cnt,1)) + LayerNorm(g1,be1), in place to hln,
// and emit the bf16 hi/lo split of h for the WMMA A-matrix.
// One block (256 threads) per segment row; 2 elements per thread.
// ---------------------------------------------------------------------------
__global__ void mean_ln_split_kernel(float* __restrict__ hln,     // in: sums, out: LN(h)
                                     const float* __restrict__ cnt,
                                     const float* __restrict__ gamma,
                                     const float* __restrict__ beta,
                                     unsigned short* __restrict__ Hhi,
                                     unsigned short* __restrict__ Hlo) {
  int row = blockIdx.x;
  int t = threadIdx.x;
  long long base = (long long)row * DDIM;
  float inv = 1.0f / fmaxf(cnt[row], 1.0f);
  float x0 = hln[base + t] * inv;
  float x1 = hln[base + 256 + t] * inv;

  __shared__ float ssum[256], ssq[256];
  ssum[t] = x0 + x1;
  ssq[t] = x0 * x0 + x1 * x1;
  __syncthreads();
  for (int off = 128; off > 0; off >>= 1) {
    if (t < off) { ssum[t] += ssum[t + off]; ssq[t] += ssq[t + off]; }
    __syncthreads();
  }
  float mu = ssum[0] * (1.0f / DDIM);
  float var = ssq[0] * (1.0f / DDIM) - mu * mu;
  float rs = rsqrtf(var + LNEPS);

  float y0 = (x0 - mu) * rs * gamma[t] + beta[t];
  float y1 = (x1 - mu) * rs * gamma[256 + t] + beta[256 + t];
  hln[base + t] = y0;
  hln[base + 256 + t] = y1;
  unsigned short h0 = f32_to_bf16_rne(y0);
  unsigned short h1 = f32_to_bf16_rne(y1);
  Hhi[base + t] = h0;
  Hhi[base + 256 + t] = h1;
  Hlo[base + t] = f32_to_bf16_rne(y0 - bf16_to_f32(h0));
  Hlo[base + 256 + t] = f32_to_bf16_rne(y1 - bf16_to_f32(h1));
}

// ---------------------------------------------------------------------------
// WMMA fragment loaders (wave32), contiguous k order k = 16*half + s in both A and B.
// A (16x32 bf16): lane L holds row m = L%16; half = L>=16.
//   V0..3 : K pairs {0,2,4,6}+16*half -> row dwords [half*8, half*8+4)
//   V4..7 : K pairs {8,10,12,14}+16*half -> row dwords [half*8+4, half*8+8)
// i.e. two contiguous 16-byte chunks of the row -> 2x global_load_b128.
// B (32x16 bf16): lane L holds col n = L%16; Vv = K pair {2v,2v+1}+16*half.
// With fragment-major packing the 8 dwords are contiguous -> 2x b128.
// C/D (16x16 f32): VGPR r -> row r + 8*half, col = L%16.
// ---------------------------------------------------------------------------
union FragU { v16bf v; uint4 q[2]; };

__device__ __forceinline__ v16bf load_a_frag(const unsigned short* __restrict__ A,
                                             long long m, int k0, int half) {
  FragU f;
  const uint4* p = (const uint4*)(A + m * DDIM + k0);
  f.q[0] = p[2 * half];      // dwords half*8 .. half*8+3   (K 0..7 | 16..23)
  f.q[1] = p[2 * half + 1];  // dwords half*8+4 ..          (K 8..15 | 24..31)
  return f.v;
}

__device__ __forceinline__ v16bf load_b_frag(const unsigned int* __restrict__ Bf,
                                             int kt, int j, int half) {
  FragU f;
  const uint4* p = (const uint4*)(Bf + ((long long)kt * DDIM + j) * 16 + half * 8);
  f.q[0] = p[0];
  f.q[1] = p[1];
  return f.v;
}

__device__ __forceinline__ v8f wmma_bf16(v16bf a, v16bf b, v8f c) {
  return __builtin_amdgcn_wmma_f32_16x16x32_bf16(false, a, false, b,
                                                 (short)0, c, false, false);
}

// ---------------------------------------------------------------------------
// Kernel 5/6: GEMM  out[s,j] = sum_k A[s,k] * W[j,k] + bias[j]  via
// compensated bf16 WMMA. Each wave: one 16x64 tile (4 accumulators),
// block = 8 waves = 8 consecutive M tiles; gridDim.y = 8 N-groups of 64.
// do_gelu => exact GELU (erf). Optional f32 output and/or bf16 hi/lo split.
// ---------------------------------------------------------------------------
__global__ void gemm_kernel(const unsigned short* __restrict__ Ahi,
                            const unsigned short* __restrict__ Alo,
                            const unsigned int* __restrict__ Bfhi,
                            const unsigned int* __restrict__ Bflo,
                            const float* __restrict__ bias,
                            float* __restrict__ outf,
                            unsigned short* __restrict__ Ohi,
                            unsigned short* __restrict__ Olo,
                            int do_gelu) {
  int wave = threadIdx.x >> 5;
  int lane = threadIdx.x & 31;
  int mtile = blockIdx.x * 8 + wave;
  if (mtile * 16 >= SEGS) return;
  int n0 = blockIdx.y * 64;
  int half = lane >> 4;
  int ml = lane & 15;
  long long m = (long long)mtile * 16 + ml;

  v8f acc[4];
#pragma unroll
  for (int t = 0; t < 4; ++t) acc[t] = (v8f){0, 0, 0, 0, 0, 0, 0, 0};

  for (int kt = 0; kt < DDIM / 32; ++kt) {
    int k0 = kt * 32;
    v16bf ah = load_a_frag(Ahi, m, k0, half);
    v16bf al = load_a_frag(Alo, m, k0, half);
#pragma unroll
    for (int t = 0; t < 4; ++t) {
      int j = n0 + t * 16 + ml;
      v16bf bh = load_b_frag(Bfhi, kt, j, half);
      v16bf bl = load_b_frag(Bflo, kt, j, half);
      acc[t] = wmma_bf16(ah, bh, acc[t]);  // hi*hi
      acc[t] = wmma_bf16(ah, bl, acc[t]);  // hi*lo  (compensation)
      acc[t] = wmma_bf16(al, bh, acc[t]);  // lo*hi  (compensation)
    }
  }

#pragma unroll
  for (int t = 0; t < 4; ++t) {
    int col = n0 + t * 16 + ml;
    float bv = bias[col];
#pragma unroll
    for (int r = 0; r < 8; ++r) {
      long long row = (long long)mtile * 16 + r + half * 8;
      float x = acc[t][r] + bv;
      if (do_gelu) x = 0.5f * x * (1.0f + erff(x * 0.70710678118654752f));
      long long idx = row * DDIM + col;
      if (outf) outf[idx] = x;
      if (Ohi) {
        unsigned short h = f32_to_bf16_rne(x);
        Ohi[idx] = h;
        Olo[idx] = f32_to_bf16_rne(x - bf16_to_f32(h));
      }
    }
  }
}

// ---------------------------------------------------------------------------
// Kernel 7: out = LayerNorm(hln + y; g2, be2)
// ---------------------------------------------------------------------------
__global__ void final_ln_kernel(const float* __restrict__ hln,
                                const float* __restrict__ y,
                                const float* __restrict__ gamma,
                                const float* __restrict__ beta,
                                float* __restrict__ out) {
  int row = blockIdx.x;
  int t = threadIdx.x;
  long long base = (long long)row * DDIM;
  float x0 = hln[base + t] + y[base + t];
  float x1 = hln[base + 256 + t] + y[base + 256 + t];

  __shared__ float ssum[256], ssq[256];
  ssum[t] = x0 + x1;
  ssq[t] = x0 * x0 + x1 * x1;
  __syncthreads();
  for (int off = 128; off > 0; off >>= 1) {
    if (t < off) { ssum[t] += ssum[t + off]; ssq[t] += ssq[t + off]; }
    __syncthreads();
  }
  float mu = ssum[0] * (1.0f / DDIM);
  float var = ssq[0] * (1.0f / DDIM) - mu * mu;
  float rs = rsqrtf(var + LNEPS);

  out[base + t] = (x0 - mu) * rs * gamma[t] + beta[t];
  out[base + 256 + t] = (x1 - mu) * rs * gamma[256 + t] + beta[256 + t];
}

// ---------------------------------------------------------------------------
// Launcher
// ---------------------------------------------------------------------------
extern "C" void kernel_launch(void* const* d_in, const int* in_sizes, int n_in,
                              void* d_out, int out_size, void* d_ws, size_t ws_size,
                              hipStream_t stream) {
  const float* edge_graph = (const float*)d_in[0];
  const int*   edge_index = (const int*)d_in[1];   // int32 (JAX x64 disabled)
  // d_in[2] = graph_index: only its length (SEGS) matters
  const float* g1  = (const float*)d_in[3];
  const float* be1 = (const float*)d_in[4];
  const float* W1  = (const float*)d_in[5];
  const float* b1  = (const float*)d_in[6];
  const float* W2  = (const float*)d_in[7];
  const float* b2  = (const float*)d_in[8];
  const float* g2  = (const float*)d_in[9];
  const float* be2 = (const float*)d_in[10];
  float* out = (float*)d_out;

  const long long SD = (long long)SEGS * DDIM;     // 25.6M elements
  char* ws = (char*)d_ws;
  size_t off = 0;
  auto take = [&](size_t bytes) { char* p = ws + off; off = (off + bytes + 255) & ~(size_t)255; return p; };

  float*          hln   = (float*)take(SD * 4);          // sums -> LN(h)
  float*          yres  = (float*)take(SD * 4);          // GEMM2 output
  unsigned short* Hhi   = (unsigned short*)take(SD * 2); // h split hi
  unsigned short* Hlo   = (unsigned short*)take(SD * 2); // h split lo
  unsigned short* Zhi   = (unsigned short*)take(SD * 2); // z split hi
  unsigned short* Zlo   = (unsigned short*)take(SD * 2); // z split lo
  float*          cnt   = (float*)take(SEGS * 4);
  unsigned int*   W1fhi = (unsigned int*)take((DDIM / 2) * DDIM * 4);
  unsigned int*   W1flo = (unsigned int*)take((DDIM / 2) * DDIM * 4);
  unsigned int*   W2fhi = (unsigned int*)take((DDIM / 2) * DDIM * 4);
  unsigned int*   W2flo = (unsigned int*)take((DDIM / 2) * DDIM * 4);

  // 1) zero accumulators
  {
    long long n = SD + SEGS;
    int blocks = (int)((n + 255) / 256);
    zero_kernel<<<blocks, 256, 0, stream>>>(hln, cnt);
  }
  // 2) weight packing (tiny, L2-resident thereafter)
  {
    int n = (DDIM / 2) * DDIM;
    int blocks = (n + 255) / 256;
    pack_w_kernel<<<blocks, 256, 0, stream>>>(W1, W1fhi, W1flo);
    pack_w_kernel<<<blocks, 256, 0, stream>>>(W2, W2fhi, W2flo);
  }
  // 3) scatter-mean accumulation (bandwidth bound: ~1.02 GB of edge reads)
  {
    long long n = (long long)EDGES * 128;
    int blocks = (int)((n + 255) / 256);
    scatter_kernel<<<blocks, 256, 0, stream>>>(edge_graph, edge_index, hln, cnt);
  }
  // 4) mean + LN1 + bf16 split
  mean_ln_split_kernel<<<SEGS, 256, 0, stream>>>(hln, cnt, g1, be1, Hhi, Hlo);
  // 5) z = gelu(h @ W1^T + b1), emitted as bf16 hi/lo split
  {
    dim3 grid((SEGS / 16 + 7) / 8, DDIM / 64);
    gemm_kernel<<<grid, 256, 0, stream>>>(Hhi, Hlo, W1fhi, W1flo, b1,
                                          nullptr, Zhi, Zlo, /*gelu=*/1);
    // 6) y = z @ W2^T + b2, f32
    gemm_kernel<<<grid, 256, 0, stream>>>(Zhi, Zlo, W2fhi, W2flo, b2,
                                          yres, nullptr, nullptr, /*gelu=*/0);
  }
  // 7) out = LN(h + y; g2, be2)
  final_ln_kernel<<<SEGS, 256, 0, stream>>>(hln, yres, g2, be2, out);
}
